// Generator_3848290697531
// MI455X (gfx1250) — matrix operand order repaired and checked by reading, not compiled, not run on hardware
//
#include <hip/hip_runtime.h>

#define BATCH 2048
#define TSTEPS 64
#define HID 64
#define NOISE 8
#define MLPW 256
#define INITD 16
#define DATA 8

typedef __attribute__((ext_vector_type(16))) _Float16 v16h;
typedef __attribute__((ext_vector_type(8)))  _Float16 v8h;
typedef __attribute__((ext_vector_type(8)))  float    v8f;

// ---------------------------------------------------------------------------
// WMMA fragment helpers (layouts per CDNA5 ISA §7.12.2)
// ---------------------------------------------------------------------------

// A-matrix (16x32 f16) fragment from a row-major LDS tile.
// lane l<16 : row=l,    K = kbase + {0..7} (VGPR0-3) and kbase + {16..23} (VGPR4-7)
// lane l>=16: row=l-16, K = kbase + {8..15}          and kbase + {24..31}
__device__ __forceinline__ v16h load_A_lds(const _Float16* alds, int ldk, int kbase, int lane) {
  const _Float16* p = alds + (lane & 15) * ldk + kbase + ((lane >> 4) << 3);
  v8h lo = *(const v8h*)(p);        // ds_load_b128
  v8h hi = *(const v8h*)(p + 16);   // ds_load_b128
  return __builtin_shufflevector(lo, hi, 0,1,2,3,4,5,6,7,8,9,10,11,12,13,14,15);
}

// B-matrix (32x16 f16) fragment, pre-packed per-lane contiguous:
// dst[(((kt*NT+nt)*32+lane)*16)+i] holds W[kt*32 + ((i>=8)?16:0) + ((lane>=16)?8:0) + (i&7)][nt*16 + (lane&15)]
__device__ __forceinline__ v16h load_B_pk(const _Float16* wp, int kt, int nt, int NT, int lane) {
  return *(const v16h*)(wp + ((((size_t)kt * NT + nt) * 32 + lane) << 4));
}

__device__ __forceinline__ v8f wmma_acc(v16h a, v16h b, v8f c) {
  return __builtin_amdgcn_wmma_f32_16x16x32_f16(false, a, false, b, (short)0, c, false, false);
}

// Fast activations: single-op v_rcp_f32 / v_tanh_f32 instead of IEEE div chains.
__device__ __forceinline__ float fast_rcp(float x) { return __builtin_amdgcn_rcpf(x); }

__device__ __forceinline__ float lipswish(float x) {
  return 0.909f * x * fast_rcp(1.0f + __expf(-x));
}

__device__ __forceinline__ float fast_tanh(float x) {
#if __has_builtin(__builtin_amdgcn_tanhf)
  return __builtin_amdgcn_tanhf(x);
#else
  const float e = __expf(2.0f * x);
  return 1.0f - 2.0f * fast_rcp(e + 1.0f);
#endif
}

// ---------------------------------------------------------------------------
// Weight pre-pack: f32 (K_src x N, optional leading-row skip, zero-pad to K)
// into the per-lane WMMA-B fragment layout above.
// ---------------------------------------------------------------------------
__global__ void pack_b_kernel(const float* __restrict__ W, _Float16* __restrict__ dst,
                              int K, int N, int srcK, int rowOff) {
  const int NT = N >> 4;
  const int total = K * N;
  for (int t = blockIdx.x * blockDim.x + threadIdx.x; t < total; t += gridDim.x * blockDim.x) {
    const int i    = t & 15;
    const int lane = (t >> 4) & 31;
    const int nt   = (t >> 9) % NT;
    const int kt   = t / (NT << 9);
    const int k = kt * 32 + ((i >> 3) << 4) + ((lane >> 4) << 3) + (i & 7);   // dense 16-bit B layout
    const int n = (nt << 4) + (lane & 15);
    const float v = (k < srcK) ? W[(size_t)(rowOff + k) * N + n] : 0.0f;
    dst[t] = (_Float16)v;
  }
}

// ---------------------------------------------------------------------------
// y0 = init-MLP(init_noise): 16(pad32) -> 256 -> 256 -> 64, lipswish x2.
// Block = 16 batch rows x 256 threads (8 waves, wave32). Wave id kept in an
// SGPR (readfirstlane) so all tile loops are scalar-controlled (no EXEC masks).
// ---------------------------------------------------------------------------
__global__ __launch_bounds__(256) void sde_init_kernel(
    const float* __restrict__ z,
    const float* __restrict__ ib0, const float* __restrict__ ib1, const float* __restrict__ ib2,
    const _Float16* __restrict__ wi0, const _Float16* __restrict__ wi1, const _Float16* __restrict__ wi2,
    float* __restrict__ ys, float* __restrict__ yhs) {
  __shared__ _Float16 a_in[16 * 32];
  __shared__ _Float16 h1[16 * MLPW];
  __shared__ _Float16 h2[16 * MLPW];

  const int tid  = threadIdx.x;
  const int lane = tid & 31;
  const int wave = __builtin_amdgcn_readfirstlane(tid >> 5);   // SGPR-uniform
  const int m0 = blockIdx.x * 16;
  const int colL = lane & 15;
  const int rbase = (lane >> 4) << 3;

  for (int e = tid; e < 16 * 32; e += 256) {
    int r = e >> 5, k = e & 31;
    a_in[e] = (_Float16)((k < INITD) ? z[(size_t)(m0 + r) * INITD + k] : 0.0f);
  }
  __syncthreads();

  {                                                  // layer 1: K=32(pad), N=256
    v16h a0 = load_A_lds(a_in, 32, 0, lane);
#pragma unroll
    for (int t2 = 0; t2 < 2; ++t2) {
      const int nt = wave * 2 + t2;
      v8f acc = {};
      acc = wmma_acc(a0, load_B_pk(wi0, 0, nt, 16, lane), acc);
      const int col = nt * 16 + colL;
      const float bias = ib0[col];
#pragma unroll
      for (int j = 0; j < 8; ++j)
        h1[(j + rbase) * MLPW + col] = (_Float16)lipswish(acc[j] + bias);
    }
  }
  __syncthreads();

  {                                                  // layer 2: K=256, N=256
    v16h af[8];
#pragma unroll
    for (int kt = 0; kt < 8; ++kt) af[kt] = load_A_lds(h1, MLPW, kt * 32, lane);
#pragma unroll
    for (int t2 = 0; t2 < 2; ++t2) {
      const int nt = wave * 2 + t2;
      v8f acc = {};
#pragma unroll
      for (int kt = 0; kt < 8; ++kt)
        acc = wmma_acc(af[kt], load_B_pk(wi1, kt, nt, 16, lane), acc);
      const int col = nt * 16 + colL;
      const float bias = ib1[col];
#pragma unroll
      for (int j = 0; j < 8; ++j)
        h2[(j + rbase) * MLPW + col] = (_Float16)lipswish(acc[j] + bias);
    }
  }
  __syncthreads();

  if (wave < 4) {                                    // layer 3: K=256, N=64 (no act)
    v16h af[8];
#pragma unroll
    for (int kt = 0; kt < 8; ++kt) af[kt] = load_A_lds(h2, MLPW, kt * 32, lane);
    const int nt = wave;
    v8f acc = {};
#pragma unroll
    for (int kt = 0; kt < 8; ++kt)
      acc = wmma_acc(af[kt], load_B_pk(wi2, kt, nt, 4, lane), acc);
    const int col = nt * 16 + colL;
    const float bias = ib2[col];
#pragma unroll
    for (int j = 0; j < 8; ++j) {
      const float v = acc[j] + bias;
      const size_t b = (size_t)(m0 + j + rbase);
      ys[b * HID + col]  = v;
      yhs[b * HID + col] = v;
    }
  }
}

// ---------------------------------------------------------------------------
// One reversible-Heun step, fully fused:
//   mode 0: evaluate (f0,g0) at (ts[0], y0), write states + readout row t=0
//   mode 1: yh1 = 2y - yh + f*dt + g.dw; (f1,g1)=MLPs(t1,yh1);
//           y1 = y + 0.5(f+f1)dt + 0.5(g+g1).dw; states <- (y1,yh1,f1,g1);
//           readout row t=tindex.
// Block = 16 batch rows, 256 threads (8 waves, wave32). All GEMMs via
// v_wmma_f32_16x16x32_f16; A fragments hoisted in VGPRs, B streamed from L2.
// ---------------------------------------------------------------------------
__global__ __launch_bounds__(256) void sde_step_kernel(
    const float* __restrict__ ts, const float* __restrict__ bm,
    const float* __restrict__ fW0, const float* __restrict__ fb0,
    const float* __restrict__ fb1, const float* __restrict__ fb2,
    const float* __restrict__ gW0, const float* __restrict__ gb0,
    const float* __restrict__ gb1, const float* __restrict__ gb2,
    const _Float16* __restrict__ wf0, const _Float16* __restrict__ wf1,
    const _Float16* __restrict__ wf2, const _Float16* __restrict__ wg0,
    const _Float16* __restrict__ wg1, const _Float16* __restrict__ wg2,
    const float* __restrict__ roW, const float* __restrict__ rob,
    float* __restrict__ ys, float* __restrict__ yhs,
    float* __restrict__ fss, float* __restrict__ gss,
    float* __restrict__ out, int tindex, int mode) {
  __shared__ _Float16 a_x[16 * HID];
  __shared__ _Float16 h_f[16 * MLPW];
  __shared__ _Float16 h_g[16 * MLPW];
  __shared__ _Float16 h_f2[16 * MLPW];
  __shared__ _Float16 h_g2[16 * MLPW];
  __shared__ float xf32[16 * HID];
  __shared__ float f1s[16 * HID];
  __shared__ float g1s[16 * HID * NOISE];
  __shared__ float bf0e[MLPW];
  __shared__ float bg0e[MLPW];

  const int tid  = threadIdx.x;
  const int lane = tid & 31;
  const int wave = __builtin_amdgcn_readfirstlane(tid >> 5);   // SGPR-uniform
  const int m0   = blockIdx.x * 16;
  const int colL = lane & 15;
  const int rbase = (lane >> 4) << 3;

  const float t1 = ts[tindex];
  const float dt = (mode == 1) ? (t1 - ts[tindex - 1]) : 0.0f;
  const float* dwp = bm;
  if (mode == 1) dwp = bm + (size_t)(tindex - 1) * (BATCH * NOISE);

  // Phase 0: fold t into layer-1 bias; build yh1 (mode1) or stage y0 (mode0).
  {
    const int n = tid;                                 // 256 threads == MLPW
    bf0e[n] = fb0[n] + t1 * fW0[n];                    // fW0 row 0 = time row
    bg0e[n] = gb0[n] + t1 * gW0[n];
  }
  for (int e = tid; e < 16 * HID; e += 256) {
    const int r = e >> 6, h = e & (HID - 1);
    const size_t b = (size_t)(m0 + r);
    float v;
    if (mode == 1) {
      float gdw = 0.0f;
      const float* gp  = gss + b * (HID * NOISE) + h * NOISE;
      const float* dwb = dwp + b * NOISE;
#pragma unroll
      for (int n = 0; n < NOISE; ++n) gdw += gp[n] * dwb[n];
      v = 2.0f * ys[b * HID + h] - yhs[b * HID + h] + fss[b * HID + h] * dt + gdw;
      yhs[b * HID + h] = v;
    } else {
      v = ys[b * HID + h];
    }
    xf32[e] = v;
    a_x[e]  = (_Float16)v;
  }
  __syncthreads();

  const bool wfp = (wave < 4);         // scalar f/g split for phases 1-2
  const int nt0 = (wave & 3) * 4;      // 4 N-tiles per wave (N=256 -> 16 tiles)

  // Phase 1: layer 1 (K=64). A fragments hoisted (2 chunks).
  {
    const _Float16* wp = wfp ? wf0 : wg0;
    const float* be = wfp ? bf0e : bg0e;
    _Float16* dst = wfp ? h_f : h_g;
    v16h af0 = load_A_lds(a_x, HID, 0, lane);
    v16h af1 = load_A_lds(a_x, HID, 32, lane);
#pragma unroll
    for (int t4 = 0; t4 < 4; ++t4) {
      const int nt = nt0 + t4;
      v8f acc = {};
      acc = wmma_acc(af0, load_B_pk(wp, 0, nt, 16, lane), acc);
      acc = wmma_acc(af1, load_B_pk(wp, 1, nt, 16, lane), acc);
      const int col = nt * 16 + colL;
      const float bias = be[col];
#pragma unroll
      for (int j = 0; j < 8; ++j)
        dst[(j + rbase) * MLPW + col] = (_Float16)lipswish(acc[j] + bias);
    }
  }
  __syncthreads();

  // Phase 2: layer 2 (K=256). A fragments hoisted (8 chunks = 64 VGPRs).
  {
    const _Float16* asrc = wfp ? h_f : h_g;
    const _Float16* wp = wfp ? wf1 : wg1;
    const float* bb = wfp ? fb1 : gb1;
    _Float16* dst = wfp ? h_f2 : h_g2;
    v16h af[8];
#pragma unroll
    for (int kt = 0; kt < 8; ++kt) af[kt] = load_A_lds(asrc, MLPW, kt * 32, lane);
#pragma unroll
    for (int t4 = 0; t4 < 4; ++t4) {
      const int nt = nt0 + t4;
      v8f acc = {};
#pragma unroll
      for (int kt = 0; kt < 8; ++kt)
        acc = wmma_acc(af[kt], load_B_pk(wp, kt, nt, 16, lane), acc);
      const int col = nt * 16 + colL;
      const float bias = bb[col];
#pragma unroll
      for (int j = 0; j < 8; ++j)
        dst[(j + rbase) * MLPW + col] = (_Float16)lipswish(acc[j] + bias);
    }
  }
  __syncthreads();

  // Phase 3a: g output layer (K=256, N=512; 32 tiles, 4 per wave), tanh.
  {
    v16h ag[8];
#pragma unroll
    for (int kt = 0; kt < 8; ++kt) ag[kt] = load_A_lds(h_g2, MLPW, kt * 32, lane);
#pragma unroll
    for (int t4 = 0; t4 < 4; ++t4) {
      const int nt = wave * 4 + t4;
      v8f acc = {};
#pragma unroll
      for (int kt = 0; kt < 8; ++kt)
        acc = wmma_acc(ag[kt], load_B_pk(wg2, kt, nt, 32, lane), acc);
      const int col = nt * 16 + colL;
      const float bias = gb2[col];
#pragma unroll
      for (int j = 0; j < 8; ++j)
        g1s[(j + rbase) * (HID * NOISE) + col] = fast_tanh(acc[j] + bias);
    }
  }
  // Phase 3b: f output layer (K=256, N=64; 4 tiles on waves 0-3), tanh.
  if (wave < 4) {
    v16h af[8];
#pragma unroll
    for (int kt = 0; kt < 8; ++kt) af[kt] = load_A_lds(h_f2, MLPW, kt * 32, lane);
    const int nt = wave;
    v8f acc = {};
#pragma unroll
    for (int kt = 0; kt < 8; ++kt)
      acc = wmma_acc(af[kt], load_B_pk(wf2, kt, nt, 4, lane), acc);
    const int col = nt * 16 + colL;
    const float bias = fb2[col];
#pragma unroll
    for (int j = 0; j < 8; ++j)
      f1s[(j + rbase) * HID + col] = fast_tanh(acc[j] + bias);
  }
  __syncthreads();

  // Phase 4a: Heun corrector, update y and f states (reads old g first).
  for (int e = tid; e < 16 * HID; e += 256) {
    const int r = e >> 6, h = e & (HID - 1);
    const size_t b = (size_t)(m0 + r);
    float y1;
    if (mode == 1) {
      float s = 0.0f;
      const float* gp  = gss + b * (HID * NOISE) + h * NOISE;
      const float* g1p = g1s + r * (HID * NOISE) + h * NOISE;
      const float* dwb = dwp + b * NOISE;
#pragma unroll
      for (int n = 0; n < NOISE; ++n) s += (gp[n] + g1p[n]) * dwb[n];
      y1 = ys[b * HID + h] + 0.5f * (fss[b * HID + h] + f1s[e]) * dt + 0.5f * s;
      ys[b * HID + h] = y1;
    } else {
      y1 = xf32[e];
    }
    fss[b * HID + h] = f1s[e];
    xf32[e] = y1;                   // y1 staged for readout
  }
  __syncthreads();

  // Phase 4b: commit g state + fused readout (y1 @ roW + rob) + time column.
  for (int e = tid; e < 16 * HID * NOISE; e += 256) {
    const int r = e >> 9;
    gss[(size_t)(m0 + r) * (HID * NOISE) + (e & (HID * NOISE - 1))] = g1s[e];
  }
  for (int e = tid; e < 16 * DATA; e += 256) {
    const int r = e >> 3, d = e & (DATA - 1);
    const size_t b = (size_t)(m0 + r);
    float s = rob[d];
#pragma unroll 8
    for (int h = 0; h < HID; ++h) s += xf32[r * HID + h] * roW[h * DATA + d];
    out[(b * TSTEPS + tindex) * (1 + DATA) + 1 + d] = s;
  }
  if (tid < 16) {
    const size_t b = (size_t)(m0 + tid);
    out[(b * TSTEPS + tindex) * (1 + DATA)] = t1;
  }
}

// ---------------------------------------------------------------------------
extern "C" void kernel_launch(void* const* d_in, const int* in_sizes, int n_in,
                              void* d_out, int out_size, void* d_ws, size_t ws_size,
                              hipStream_t stream) {
  (void)in_sizes; (void)n_in; (void)out_size; (void)ws_size;
  const float* ts  = (const float*)d_in[0];
  const float* z   = (const float*)d_in[2];
  const float* bm  = (const float*)d_in[3];
  const float* iW0 = (const float*)d_in[4];  const float* ib0 = (const float*)d_in[5];
  const float* iW1 = (const float*)d_in[6];  const float* ib1 = (const float*)d_in[7];
  const float* iW2 = (const float*)d_in[8];  const float* ib2 = (const float*)d_in[9];
  const float* fW0 = (const float*)d_in[10]; const float* fb0 = (const float*)d_in[11];
  const float* fW1 = (const float*)d_in[12]; const float* fb1 = (const float*)d_in[13];
  const float* fW2 = (const float*)d_in[14]; const float* fb2 = (const float*)d_in[15];
  const float* gW0 = (const float*)d_in[16]; const float* gb0 = (const float*)d_in[17];
  const float* gW1 = (const float*)d_in[18]; const float* gb1 = (const float*)d_in[19];
  const float* gW2 = (const float*)d_in[20]; const float* gb2 = (const float*)d_in[21];
  const float* roW = (const float*)d_in[22]; const float* rob = (const float*)d_in[23];
  float* out = (float*)d_out;

  // Workspace carve-out (packed f16 weights + recurrent state, ~6.3 MB).
  char* ws = (char*)d_ws;
  size_t off = 0;
  auto bump = [&](size_t bytes) -> void* {
    void* p = (void*)(ws + off);
    off = (off + bytes + 255) & ~(size_t)255;
    return p;
  };
  _Float16* wi0 = (_Float16*)bump((size_t)32  * 256 * 2);
  _Float16* wi1 = (_Float16*)bump((size_t)256 * 256 * 2);
  _Float16* wi2 = (_Float16*)bump((size_t)256 * 64  * 2);
  _Float16* wf0 = (_Float16*)bump((size_t)64  * 256 * 2);
  _Float16* wf1 = (_Float16*)bump((size_t)256 * 256 * 2);
  _Float16* wf2 = (_Float16*)bump((size_t)256 * 64  * 2);
  _Float16* wg0 = (_Float16*)bump((size_t)64  * 256 * 2);
  _Float16* wg1 = (_Float16*)bump((size_t)256 * 256 * 2);
  _Float16* wg2 = (_Float16*)bump((size_t)256 * 512 * 2);
  float* ys  = (float*)bump((size_t)BATCH * HID * 4);
  float* yhs = (float*)bump((size_t)BATCH * HID * 4);
  float* fss = (float*)bump((size_t)BATCH * HID * 4);
  float* gss = (float*)bump((size_t)BATCH * HID * NOISE * 4);

  auto pack = [&](const float* W, _Float16* dst, int K, int N, int srcK, int rowOff) {
    const int total = K * N;
    pack_b_kernel<<<(total + 255) / 256, 256, 0, stream>>>(W, dst, K, N, srcK, rowOff);
  };
  pack(iW0, wi0, 32,  256, 16,  0);   // init layer 1 (K padded 16->32)
  pack(iW1, wi1, 256, 256, 256, 0);
  pack(iW2, wi2, 256, 64,  256, 0);
  pack(fW0, wf0, 64,  256, 64,  1);   // skip time row (folded into bias)
  pack(fW1, wf1, 256, 256, 256, 0);
  pack(fW2, wf2, 256, 64,  256, 0);
  pack(gW0, wg0, 64,  256, 64,  1);   // skip time row
  pack(gW1, wg1, 256, 256, 256, 0);
  pack(gW2, wg2, 256, 512, 256, 0);

  const dim3 grid(BATCH / 16), blk(256);
  sde_init_kernel<<<grid, blk, 0, stream>>>(z, ib0, ib1, ib2, wi0, wi1, wi2, ys, yhs);

  // t=0: evaluate (f0, g0) and emit readout row 0.
  sde_step_kernel<<<grid, blk, 0, stream>>>(ts, bm, fW0, fb0, fb1, fb2, gW0, gb0, gb1, gb2,
                                            wf0, wf1, wf2, wg0, wg1, wg2, roW, rob,
                                            ys, yhs, fss, gss, out, 0, 0);
  // Sequential scan: 63 reversible-Heun steps.
  for (int i = 1; i < TSTEPS; ++i)
    sde_step_kernel<<<grid, blk, 0, stream>>>(ts, bm, fW0, fb0, fb1, fb2, gW0, gb0, gb1, gb2,
                                              wf0, wf1, wf2, wg0, wg1, wg2, roW, rob,
                                              ys, yhs, fss, gss, out, i, 1);
}
